// CPCNet_3092376453316
// MI455X (gfx1250) — matrix operand order repaired and checked by reading, not compiled, not run on hardware
//
#include <hip/hip_runtime.h>

// ---------------------------------------------------------------------------
// CPC net forward for MI455X (gfx1250), wave32, WMMA bf16 path.
//  k_prep  : W_embed -> bf16, transposed, tiled exactly as the LDS B layout
//  k_embed : [24576 x 8400] @ [8400 x 100] + bias -> Ec, Xp_full  (dominant;
//            double-buffered LDS, register-pipelined A stage, async B copy)
//  k_gi    : Ec @ W_ih^T + b_ih -> gi[2048 x 300]
//  k_gru   : persistent 1-WG GRU, h/gh in LDS, WMMA for h@W_hh^T
//  k_bilW  : per-s hidden @ W_bil[s] -> M[b,s,e]
//  k_dot   : out[b,s,p] = dot(M[b,s,:], Xp_full[b,s,p,:])
// ---------------------------------------------------------------------------

typedef __attribute__((ext_vector_type(16))) __bf16 v16bf;
typedef __attribute__((ext_vector_type(2)))  __bf16 v2bf;
typedef __attribute__((ext_vector_type(8)))  float  v8f;

union FragU { v16bf v; unsigned int u[8]; };

__device__ __forceinline__ unsigned short f2bf(float f) {
  unsigned int x = __float_as_uint(f);
  x += 0x7FFFu + ((x >> 16) & 1u);          // round-to-nearest-even
  return (unsigned short)(x >> 16);
}
// Packed f32x2 -> bf16x2 (single v_cvt_pk_bf16_f32 when available).
__device__ __forceinline__ unsigned int pack2bf(float lo, float hi) {
#if __has_builtin(__builtin_amdgcn_cvt_pk_bf16_f32)
  union { v2bf v; unsigned int u; } c;
  c.v = __builtin_amdgcn_cvt_pk_bf16_f32(lo, hi);
  return c.u;
#else
  return (unsigned int)f2bf(lo) | ((unsigned int)f2bf(hi) << 16);
#endif
}

// CDNA5 async global->LDS copy (16B per lane) + its counter wait.
// VDST vgpr = workgroup-relative LDS byte offset (low 32 bits of a generic
// pointer to __shared__), VADDR = 64-bit global address. Tracked by ASYNCcnt.
__device__ __forceinline__ void async_ld_b128(unsigned lds_off, const void* g) {
  asm volatile("global_load_async_to_lds_b128 %0, %1, off"
               :: "v"(lds_off), "v"(g) : "memory");
}
__device__ __forceinline__ void wait_async0() {
  asm volatile("s_wait_asynccnt 0x0" ::: "memory");
}

// A fragment: 16x32 bf16, rows in lanes (row = lane%16), K striped per ISA:
// g=lane/16: VGPR j<4 -> K=8g+2j,+1 ; j>=4 -> K=16+8g+2(j-4),+1
__device__ __forceinline__ v16bf fragA_lds(const unsigned short* base, int lane, int ld) {
  int row = lane & 15, g = lane >> 4;
  const unsigned short* p = base + row * ld;
  FragU f;
#pragma unroll
  for (int j = 0; j < 8; ++j) {
    int k = (j < 4) ? (8 * g + 2 * j) : (16 + 8 * g + 2 * (j - 4));
    f.u[j] = *(const unsigned int*)(p + k);   // k even -> 4B aligned
  }
  return f.v;
}
// B fragment: 32x16 bf16, cols in lanes (col = lane%16), K striped as A: j<4 -> 8g+2j,+1 ; j>=4 -> 16+8g+2(j-4),+1.
// LDS holds B transposed: [n][k], pairs (k,k+1) contiguous.
__device__ __forceinline__ v16bf fragBT_lds(const unsigned short* baseT, int lane, int ld) {
  int col = lane & 15, g = lane >> 4;
  const unsigned short* p = baseT + col * ld;
  FragU f;
#pragma unroll
  for (int j = 0; j < 8; ++j)
    f.u[j] = *(const unsigned int*)(p + ((j < 4) ? (8 * g + 2 * j) : (16 + 8 * g + 2 * (j - 4))));
  return f.v;
}

namespace cfg {
constexpr int    Bn = 128, NE = 16, NB = 10, E = 100, H = 100;
constexpr int    KE = 21 * 400;                                  // 8400
constexpr int    P  = NB + 1;                                    // 11
constexpr int    KBLK = (KE + 31) / 32;                          // 263
// workspace offsets in floats
constexpr size_t OFF_EC  = 0;                                    // [2048 x 100]
constexpr size_t OFF_XPF = OFF_EC  + (size_t)Bn * NE * E;        // [B,16,11,100]
constexpr size_t OFF_GI  = OFF_XPF + (size_t)Bn * NE * P * E;    // [2048 x 300]
constexpr size_t OFF_HID = OFF_GI  + (size_t)Bn * NE * 3 * H;    // [128 x 100]
constexpr size_t OFF_M   = OFF_HID + (size_t)Bn * H;             // [B,16,100]
constexpr size_t OFF_WT  = OFF_M   + (size_t)Bn * NE * E;        // ushort[263*3584]
}

// ---------------------------------------------------------------------------
// Kernel 0: W_embed -> WT[kblk][n][kk] bf16 (exact LDS B-tile image, zero-pad)
// ---------------------------------------------------------------------------
__global__ __launch_bounds__(256) void k_prep(
    const float* __restrict__ Wemb, float* __restrict__ ws)
{
  unsigned short* WT = (unsigned short*)(ws + cfg::OFF_WT);
  int kb = blockIdx.x, k0 = kb * 32;
  for (int i = threadIdx.x; i < 112 * 16; i += 256) {   // packed pairs over kk
    int n = i >> 4, kk = 2 * (i & 15);
    int k = k0 + kk;
    float lo = 0.f, hi = 0.f;
    if (n < 100 && k < cfg::KE)     lo = Wemb[(size_t)k * 100 + n];
    if (n < 100 && k + 1 < cfg::KE) hi = Wemb[(size_t)(k + 1) * 100 + n];
    *(unsigned int*)(WT + (size_t)kb * 3584 + n * 32 + kk) = pack2bf(lo, hi);
  }
}

// ---------------------------------------------------------------------------
// Kernel 1: embed GEMM. 192 blocks x 256 thr. Block = 128 rows x 112 cols.
// Double-buffered: A staged via float4 regs -> packed bf16 uint2 LDS stores,
// B tile copied with async global->LDS b128. One barrier per k-step.
// ---------------------------------------------------------------------------
__global__ __launch_bounds__(256) void k_embed(
    const float* __restrict__ Xc, const float* __restrict__ Xp,
    const float* __restrict__ Xb, const float* __restrict__ bemb,
    float* __restrict__ ws)
{
  __shared__ const float* rsrc[128];
  __shared__ float*       rdst[128];
  __shared__ __align__(16) unsigned short lA[2][128 * 32];
  __shared__ __align__(16) unsigned short lB[2][112 * 32];
  float* Ec  = ws + cfg::OFF_EC;
  float* Xpf = ws + cfg::OFF_XPF;
  const unsigned short* WT = (const unsigned short*)(ws + cfg::OFF_WT);
  int tid = threadIdx.x;

  if (tid < 128) {
    int r = blockIdx.x * 128 + tid;
    const float* s; float* d;
    if (r < 2048) {                       // Xc rows -> Ec
      s = Xc + (size_t)r * cfg::KE;  d = Ec + (size_t)r * 100;
    } else if (r < 4096) {                // Xp rows -> Xp_full p=0
      int q = r - 2048;
      s = Xp + (size_t)q * cfg::KE;  d = Xpf + (size_t)q * 11 * 100;
    } else {                              // Xb rows -> Xp_full p=1..10
      int q = r - 4096;                   // q = (b*16+s)*10 + p
      int bs = q / 10, p = q % 10;
      s = Xb + (size_t)q * cfg::KE;  d = Xpf + ((size_t)bs * 11 + 1 + p) * 100;
    }
    rsrc[tid] = s; rdst[tid] = d;
  }
  __syncthreads();

  int w = tid >> 5, lane = tid & 31;
  v8f vz = {0.f,0.f,0.f,0.f,0.f,0.f,0.f,0.f};
  v8f acc[7];
#pragma unroll
  for (int t = 0; t < 7; ++t) acc[t] = vz;

  // A register pipeline: thread handles 4 quads (float4) of the 128x32 tile.
  float4 areg[4];
  auto loadA = [&](int k0) {
#pragma unroll
    for (int m = 0; m < 4; ++m) {
      int i = tid + 256 * m;
      int row = i >> 3, k = k0 + 4 * (i & 7);      // quad-aligned; 8400%4==0
      float4 v = {0.f,0.f,0.f,0.f};
      if (k < cfg::KE) {
        v = *(const float4*)(rsrc[row] + k);
        if (k + 64 < cfg::KE) __builtin_prefetch(rsrc[row] + k + 64, 0, 0);
      }
      areg[m] = v;
    }
  };
  auto storeA = [&](unsigned short* dst) {
#pragma unroll
    for (int m = 0; m < 4; ++m) {
      int i = tid + 256 * m;
      int row = i >> 3, q4 = 4 * (i & 7);
      uint2 pk;
      pk.x = pack2bf(areg[m].x, areg[m].y);
      pk.y = pack2bf(areg[m].z, areg[m].w);
      *(uint2*)(dst + row * 32 + q4) = pk;         // 8B-aligned ds_store_b64
    }
  };
  auto asyncB = [&](unsigned short* dst, int kblk) {
    const unsigned short* src = WT + (size_t)kblk * 3584;   // 7168B contiguous
    unsigned base = (unsigned)(uintptr_t)dst;               // LDS byte offset
#pragma unroll
    for (int m = 0; m < 2; ++m) {
      int i = tid + 256 * m;                                // 448 x 16B chunks
      if (i < 448) async_ld_b128(base + (unsigned)i * 16u, src + i * 8);
    }
  };

  // prologue: fill buffer 0
  loadA(0);
  asyncB(lB[0], 0);
  storeA(lA[0]);
  wait_async0();
  __syncthreads();

  for (int it = 0; it < cfg::KBLK; ++it) {         // 263 k-steps
    unsigned short* curA = lA[it & 1];
    unsigned short* curB = lB[it & 1];
    unsigned short* nxtA = lA[(it & 1) ^ 1];
    unsigned short* nxtB = lB[(it & 1) ^ 1];
    bool more = (it + 1 < cfg::KBLK);
    if (more) { loadA(32 * (it + 1)); asyncB(nxtB, it + 1); }

    v16bf af = fragA_lds(curA + (16 * w) * 32, lane, 32);
    v16bf bcur = fragBT_lds(curB, lane, 32);
#pragma unroll
    for (int t = 0; t < 7; ++t) {                  // one-ahead B fragment
      v16bf bnext;
      if (t < 6) bnext = fragBT_lds(curB + (16 * (t + 1)) * 32, lane, 32);
      acc[t] = __builtin_amdgcn_wmma_f32_16x16x32_bf16(
          false, af, false, bcur, (short)0, acc[t], false, false);
      bcur = bnext;
    }
    if (more) storeA(nxtA);
    wait_async0();
    __syncthreads();
  }

  int g = lane >> 4, cl = lane & 15;
#pragma unroll
  for (int t = 0; t < 7; ++t) {
    int col = 16 * t + cl;
    if (col < 100) {
      float bias = bemb[col];
#pragma unroll
      for (int r = 0; r < 8; ++r)
        rdst[16 * w + r + 8 * g][col] = acc[t][r] + bias;
    }
  }
}

// ---------------------------------------------------------------------------
// Kernel 2: gi = Ec @ W_ih^T + b_ih.  16 blocks x 256 thr.
// M=128/block, N=300 (19 tiles; wave w -> tiles w, w+8, w+16), K=100 (4x32).
// ---------------------------------------------------------------------------
__global__ __launch_bounds__(256) void k_gi(
    const float* __restrict__ Wih, const float* __restrict__ bih,
    float* __restrict__ ws)
{
  const float* Ec = ws + cfg::OFF_EC;
  float*       gi = ws + cfg::OFF_GI;
  __shared__ __align__(16) unsigned short lA[128 * 32];
  __shared__ __align__(16) unsigned short lB[304 * 32];
  int tid = threadIdx.x, w = tid >> 5, lane = tid & 31;
  int row0 = blockIdx.x * 128;
  v8f vz = {0.f,0.f,0.f,0.f,0.f,0.f,0.f,0.f};
  v8f acc[3] = {vz, vz, vz};

  for (int ks = 0; ks < 4; ++ks) {
    int k0 = ks * 32;
    for (int i = tid; i < 128 * 16; i += 256) {     // packed-pair A stage
      int row = i >> 4, k = k0 + 2 * (i & 15);
      float lo = 0.f, hi = 0.f;
      if (k < 100) { float2 v = *(const float2*)(Ec + (size_t)(row0 + row) * 100 + k); lo = v.x; hi = v.y; }
      *(unsigned int*)(lA + row * 32 + 2 * (i & 15)) = pack2bf(lo, hi);
    }
    for (int i = tid; i < 304 * 16; i += 256) {     // B^T[n][k] = W_ih[n*100+k]
      int n = i >> 4, k = k0 + 2 * (i & 15);
      float lo = 0.f, hi = 0.f;
      if (k < 100 && n < 300) { float2 v = *(const float2*)(Wih + (size_t)n * 100 + k); lo = v.x; hi = v.y; }
      *(unsigned int*)(lB + n * 32 + 2 * (i & 15)) = pack2bf(lo, hi);
    }
    __syncthreads();
    v16bf af = fragA_lds(lA + (16 * w) * 32, lane, 32);
#pragma unroll
    for (int m = 0; m < 3; ++m) {
      int t = w + 8 * m;
      if (t < 19) {                                  // wave-uniform branch
        v16bf bf = fragBT_lds(lB + (16 * t) * 32, lane, 32);
        acc[m] = __builtin_amdgcn_wmma_f32_16x16x32_bf16(
            false, af, false, bf, (short)0, acc[m], false, false);
      }
    }
    __syncthreads();
  }

  int g = lane >> 4, cl = lane & 15;
#pragma unroll
  for (int m = 0; m < 3; ++m) {
    int t = w + 8 * m;
    if (t < 19) {
      int col = 16 * t + cl;
      if (col < 300) {
        float bias = bih[col];
#pragma unroll
        for (int r = 0; r < 8; ++r)
          gi[(size_t)(row0 + 16 * w + r + 8 * g) * 300 + col] = acc[m][r] + bias;
      }
    }
  }
}

// ---------------------------------------------------------------------------
// Kernel 3: recurrent GRU, single persistent workgroup (1024 thr = 32 waves).
// h[128x100] f32 and gh[128x304] f32 in LDS (~207 KB of 320 KB/WGP).
// ---------------------------------------------------------------------------
__global__ __launch_bounds__(1024) void k_gru(
    const float* __restrict__ Whh, const float* __restrict__ bhh,
    float* __restrict__ ws)
{
  extern __shared__ float sm[];
  float* h  = sm;                    // [128*100]
  float* gh = sm + 128 * 100;        // [128*304]
  const float* gi = ws + cfg::OFF_GI;
  float* hid = ws + cfg::OFF_HID;
  int tid = threadIdx.x, w = tid >> 5, lane = tid & 31;

  for (int i = tid; i < 128 * 100; i += 1024) h[i] = 0.f;

  for (int t = 0; t < 16; ++t) {
    __syncthreads();                                   // h stable for this step
    for (int job = w; job < 8 * 19; job += 32) {       // 152 tile jobs / 32 waves
      int mt = job / 19, nt = job % 19;
      int g = lane >> 4, cl = lane & 15;
      int arow = 16 * mt + cl;
      int col  = 16 * nt + cl;
      v8f acc = {0.f,0.f,0.f,0.f,0.f,0.f,0.f,0.f};
#pragma unroll
      for (int ks = 0; ks < 4; ++ks) {
        int k0 = ks * 32;
        FragU fa, fb;
#pragma unroll
        for (int j = 0; j < 8; ++j) {                  // A: h (f32 LDS -> bf16)
          int kk = (j < 4) ? (8 * g + 2 * j) : (16 + 8 * g + 2 * (j - 4));
          int k = k0 + kk;
          float lo = 0.f, hi = 0.f;
          if (k < 100) { float2 v = *(const float2*)(h + arow * 100 + k); lo = v.x; hi = v.y; }
          fa.u[j] = pack2bf(lo, hi);
        }
#pragma unroll
        for (int j = 0; j < 8; ++j) {                  // B: W_hh^T from L2
          int e = k0 + ((j < 4) ? (8 * g + 2 * j) : (16 + 8 * g + 2 * (j - 4)));
          float lo = 0.f, hi = 0.f;
          if (col < 300 && e < 100) { float2 v = *(const float2*)(Whh + (size_t)col * 100 + e); lo = v.x; hi = v.y; }
          fb.u[j] = pack2bf(lo, hi);
        }
        acc = __builtin_amdgcn_wmma_f32_16x16x32_bf16(
            false, fa.v, false, fb.v, (short)0, acc, false, false);
      }
#pragma unroll
      for (int r = 0; r < 8; ++r)
        gh[(16 * mt + r + 8 * (lane >> 4)) * 304 + col] = acc[r];
    }
    __syncthreads();                                   // gh complete
    for (int i = tid; i < 128 * 100; i += 1024) {      // fused gates
      int b = i / 100, j = i % 100;
      const float* gib = gi + (size_t)(b * 16 + t) * 300;
      float ir = gib[j], iz = gib[100 + j], in_ = gib[200 + j];
      float hr = gh[b * 304 + j]       + bhh[j];
      float hz = gh[b * 304 + 100 + j] + bhh[100 + j];
      float hn = gh[b * 304 + 200 + j] + bhh[200 + j];
      float r = 1.f / (1.f + __expf(-(ir + hr)));
      float z = 1.f / (1.f + __expf(-(iz + hz)));
      float n = tanhf(in_ + r * hn);
      h[i] = (1.f - z) * n + z * h[i];
    }
  }
  __syncthreads();
  for (int i = tid; i < 128 * 100; i += 1024) hid[i] = h[i];
}

// ---------------------------------------------------------------------------
// Kernel 4: M[b,s,:] = hidden @ W_bil[s].  16 blocks (one per s) x 256 thr.
// ---------------------------------------------------------------------------
__global__ __launch_bounds__(256) void k_bilW(
    const float* __restrict__ Wbil, float* __restrict__ ws)
{
  const float* hid = ws + cfg::OFF_HID;
  float*       M   = ws + cfg::OFF_M;
  int s = blockIdx.x;
  __shared__ __align__(16) unsigned short lA[128 * 32];
  __shared__ __align__(16) unsigned short lB[112 * 32];
  int tid = threadIdx.x, w = tid >> 5, lane = tid & 31;
  v8f vz = {0.f,0.f,0.f,0.f,0.f,0.f,0.f,0.f};
  v8f acc[7];
#pragma unroll
  for (int t = 0; t < 7; ++t) acc[t] = vz;

  for (int ks = 0; ks < 4; ++ks) {
    int k0 = ks * 32;
    for (int i = tid; i < 128 * 16; i += 256) {
      int row = i >> 4, k = k0 + 2 * (i & 15);
      float lo = 0.f, hi = 0.f;
      if (k < 100) { float2 v = *(const float2*)(hid + (size_t)row * 100 + k); lo = v.x; hi = v.y; }
      *(unsigned int*)(lA + row * 32 + 2 * (i & 15)) = pack2bf(lo, hi);
    }
    for (int i = tid; i < 112 * 16; i += 256) {     // B^T[n=e][k=h] = Wbil[s,h,e]
      int n = i >> 4, k = k0 + 2 * (i & 15);
      float lo = 0.f, hi = 0.f;
      if (k < 100 && n < 100) {
        lo = Wbil[((size_t)s * 100 + k) * 100 + n];
        if (k + 1 < 100) hi = Wbil[((size_t)s * 100 + k + 1) * 100 + n];
      }
      *(unsigned int*)(lB + n * 32 + 2 * (i & 15)) = pack2bf(lo, hi);
    }
    __syncthreads();
    v16bf af = fragA_lds(lA + (16 * w) * 32, lane, 32);
#pragma unroll
    for (int t = 0; t < 7; ++t) {
      v16bf bf = fragBT_lds(lB + (16 * t) * 32, lane, 32);
      acc[t] = __builtin_amdgcn_wmma_f32_16x16x32_bf16(
          false, af, false, bf, (short)0, acc[t], false, false);
    }
    __syncthreads();
  }

  int g = lane >> 4, cl = lane & 15;
#pragma unroll
  for (int t = 0; t < 7; ++t) {
    int col = 16 * t + cl;
    if (col < 100) {
#pragma unroll
      for (int r = 0; r < 8; ++r) {
        int b = 16 * w + r + 8 * g;
        M[((size_t)b * 16 + s) * 100 + col] = acc[t][r];
      }
    }
  }
}

// ---------------------------------------------------------------------------
// Kernel 5: out[b,s,p] = dot(M[b,s,:], Xp_full[b,s,p,:]).  One wave / output.
// ---------------------------------------------------------------------------
__global__ __launch_bounds__(256) void k_dot(
    float* __restrict__ out, const float* __restrict__ ws)
{
  const float* M   = ws + cfg::OFF_M;
  const float* Xpf = ws + cfg::OFF_XPF;
  int gw = blockIdx.x * 8 + (threadIdx.x >> 5);
  int lane = threadIdx.x & 31;
  if (gw >= 128 * 16 * 11) return;
  int b = gw / 176, rem = gw % 176, s = rem / 11, p = rem % 11;
  const float* m = M   + ((size_t)b * 16 + s) * 100;
  const float* x = Xpf + (((size_t)b * 16 + s) * 11 + p) * 100;
  float sum = 0.f;
  for (int e = lane; e < 100; e += 32) sum += m[e] * x[e];
#pragma unroll
  for (int off = 16; off > 0; off >>= 1) sum += __shfl_xor(sum, off, 32);
  if (lane == 0) out[gw] = sum;
}

// ---------------------------------------------------------------------------
extern "C" void kernel_launch(void* const* d_in, const int* in_sizes, int n_in,
                              void* d_out, int out_size, void* d_ws, size_t ws_size,
                              hipStream_t stream) {
  (void)in_sizes; (void)n_in; (void)out_size; (void)ws_size;
  const float* Xc   = (const float*)d_in[0];
  const float* Xp   = (const float*)d_in[1];
  const float* Xb   = (const float*)d_in[2];
  const float* Wemb = (const float*)d_in[3];
  const float* bemb = (const float*)d_in[4];
  const float* Wih  = (const float*)d_in[5];
  const float* Whh  = (const float*)d_in[6];
  const float* bih  = (const float*)d_in[7];
  const float* bhh  = (const float*)d_in[8];
  const float* Wbil = (const float*)d_in[9];
  float* out = (float*)d_out;
  float* ws  = (float*)d_ws;

  k_prep <<<cfg::KBLK, 256, 0, stream>>>(Wemb, ws);               // W -> bf16 tiles
  k_embed<<<192, 256, 0, stream>>>(Xc, Xp, Xb, bemb, ws);         // 24576 rows / 128
  k_gi   <<<16, 256, 0, stream>>>(Wih, bih, ws);                  // 2048 rows / 128
  size_t lds3 = (size_t)(128 * 100 + 128 * 304) * sizeof(float);  // 206,848 B
  k_gru  <<<1, 1024, lds3, stream>>>(Whh, bhh, ws);
  k_bilW <<<16, 256, 0, stream>>>(Wbil, ws);
  k_dot  <<<(128 * 16 * 11) / 8, 256, 0, stream>>>(out, ws);
}
